// ConvGeodesic_1563368096532
// MI455X (gfx1250) — matrix operand order repaired and checked by reading, not compiled, not run on hardware
//
#include <hip/hip_runtime.h>

// MI455X / gfx1250 implementation:
//  K1: build rotated+transposed kernel matrix Bt (f16, 256 x 1280) in ws
//  K2: barycentric pullback -> X (f16, M x 1280) in ws (LDS bin accumulate)
//  K3: WMMA f16 GEMM (M x 1280) x (1280 x 256): wave = 16 rows x 16 o-cols
//      x 8 rotations (8 accumulator banks). Double-buffered async global->LDS
//      B staging overlapped with compute; B fragment ds_loads and A chunk
//      global loads software-pipelined one step ahead. Fused relu+max.

typedef __attribute__((ext_vector_type(16))) _Float16 v16h;
typedef __attribute__((ext_vector_type(8)))  float    v8f;

// Async global->LDS builtin operand types: int4 in AS(1) / AS(3).
typedef int v4i_t __attribute__((__vector_size__(16)));
typedef v4i_t __attribute__((address_space(1)))* gptr_v4i;
typedef v4i_t __attribute__((address_space(3)))* lptr_v4i;

#define N_IN    32
#define N_OUT   32
#define N_RHO   5
#define N_THETA 8
#define KV      40
#define KDIM    1280   // KV * N_IN
#define NCOL    256    // N_THETA rotations * N_OUT

#define KSLICE    32   // K columns per LDS slice (one wmma chunk)
#define NSLICES   (KDIM / KSLICE)   // 40
#define COLSTRIDE 80   // bytes per column LDS block: 64 data + 16 pad
#define SLICEBYTES (NCOL * COLSTRIDE)   // 20480

// ---------------------------------------------------------------------------
// Kernel 1: Bt[c][q] = kernel[i][(j+r)%8][o][n]   (f32 -> f16)
//   c = r*32 + o (0..255), q = (i*8+j)*32 + n (0..1279)
// ---------------------------------------------------------------------------
__global__ void cg_build_bt(const float* __restrict__ kern,
                            _Float16* __restrict__ Bt) {
  int t = blockIdx.x * blockDim.x + threadIdx.x;
  if (t >= NCOL * KDIM) return;
  int c = t / KDIM;
  int q = t - c * KDIM;
  int r = c >> 5, o = c & 31;
  int i = q >> 8;
  int j = (q >> 5) & 7;
  int n = q & 31;
  int jr = (j + r) & 7;
  float v = kern[(((i * N_THETA) + jr) * N_OUT + o) * N_IN + n];
  Bt[t] = (_Float16)v;
}

// ---------------------------------------------------------------------------
// Kernel 2: pullback + bin scatter. One wave per vertex, 8 vertices/block.
// Lane `l` only writes LDS addresses == l (mod 32) -> race-free, no atomics.
// Writeback packs 2 f16 per dword (cross-lane LDS reads, same wave: in-order).
// ---------------------------------------------------------------------------
__global__ void cg_pullback(const float* __restrict__ sig,
                            const float* __restrict__ w,
                            const int*   __restrict__ idx,
                            const int*   __restrict__ rad,
                            const int*   __restrict__ ang,
                            _Float16*    __restrict__ X, int mv) {
  __shared__ float lds[8 * KDIM];            // 40 KB
  const int wave = threadIdx.x >> 5;
  const int lane = threadIdx.x & 31;
  const int v = blockIdx.x * 8 + wave;       // wave-uniform
  float* acc = &lds[wave * KDIM];
  for (int i = lane; i < KDIM; i += 32) acc[i] = 0.0f;
  if (v < mv) {
    const size_t base = (size_t)v * KV;
    for (int k = 0; k < KV; ++k) {
      const size_t b3 = (base + k) * 3;
      float w0 = w[b3 + 0], w1 = w[b3 + 1], w2 = w[b3 + 2];
      int   i0 = idx[b3 + 0], i1 = idx[b3 + 1], i2 = idx[b3 + 2];
      int fl = rad[base + k] * N_THETA + ang[base + k];
      fl = fl < 0 ? 0 : (fl >= KV ? KV - 1 : fl);
      float s = w0 * sig[i0 * N_IN + lane]
              + w1 * sig[i1 * N_IN + lane]
              + w2 * sig[i2 * N_IN + lane];
      acc[fl * 32 + lane] += s;
    }
    __builtin_amdgcn_wave_barrier();         // pin LDS write->read order
    unsigned int* row32 = (unsigned int*)(X + (size_t)v * KDIM);
    union HU { _Float16 h[2]; unsigned int u; };
    for (int j = 0; j < KV / 2; ++j) {       // 20 packed dword stores
      HU p;
      p.h[0] = (_Float16)acc[j * 64 + 2 * lane + 0];
      p.h[1] = (_Float16)acc[j * 64 + 2 * lane + 1];
      row32[j * 32 + lane] = p.u;
    }
  }
}

// ---------------------------------------------------------------------------
// Kernel 3 helpers
// ---------------------------------------------------------------------------
union V { uint4 q[2]; v16h h; };

__device__ __forceinline__ void cg_stage_slice(const _Float16* __restrict__ Bt,
                                               char* dst, int tid, int sbase) {
  // 1024 x 16B units per slice, 4 per thread (256 threads).
#pragma unroll
  for (int it = 0; it < 4; ++it) {
    int u = tid + (it << 8);
    int q = u & 3;              // 16B quarter within 64B column block
    int c = u >> 2;             // column 0..255
    const _Float16* g = Bt + (size_t)c * KDIM + sbase + q * 8;
    char* l = dst + c * COLSTRIDE + q * 16;
#if __has_builtin(__builtin_amdgcn_global_load_async_to_lds_b128)
    __builtin_amdgcn_global_load_async_to_lds_b128(
        (gptr_v4i)g, (lptr_v4i)l, 0, 0);
#else
    *(uint4*)l = *(const uint4*)g;
#endif
  }
}

__device__ __forceinline__ void cg_wait_async() {
#if __has_builtin(__builtin_amdgcn_global_load_async_to_lds_b128)
  asm volatile("s_wait_asynccnt 0" ::: "memory");
#endif
}

// ---------------------------------------------------------------------------
// Kernel 3: C = X * Bt^T with v_wmma_f32_16x16x32_f16.
// Wave = 16(M) x 16(o) tile x 8 rotations; waves 2t/2t+1 share an M-tile
// (o-halves) so A rows are served once from HBM then by WGP$/L2.
// ---------------------------------------------------------------------------
__global__ void __launch_bounds__(256, 1)
cg_gemm_rot_max(const _Float16* __restrict__ X,
                const _Float16* __restrict__ Bt,
                float* __restrict__ out,
                int mtiles, int mv) {
  __shared__ char bsl[2 * SLICEBYTES];              // 40960 B, ping-pong
  const int tid   = threadIdx.x;
  const int wave  = tid >> 5;
  const int lane  = tid & 31;
  const int mtile = blockIdx.x * 4 + (wave >> 1);   // wave-uniform
  const bool active = (mtile < mtiles);
  const int osub  = (wave & 1) << 4;                // 0 or 16
  const int lhalf = lane >> 4;                      // 0 or 1
  const int lmod  = lane & 15;

  const _Float16* arow =
      X + (size_t)(mtile * 16 + lmod) * KDIM + lhalf * 8;
  const int bcol = (osub + lmod) * COLSTRIDE + lhalf * 16;

  v8f zero = {0.f, 0.f, 0.f, 0.f, 0.f, 0.f, 0.f, 0.f};
  v8f acc[8];
#pragma unroll
  for (int r = 0; r < 8; ++r) acc[r] = zero;

  // Prologue: stage slice 0, preload A chunk 0.
  cg_stage_slice(Bt, bsl, tid, 0);
  V av;
  if (active) {
    av.q[0] = *(const uint4*)(arow);
    av.q[1] = *(const uint4*)(arow + 16);
  }
  cg_wait_async();
  __syncthreads();

#pragma unroll 2
  for (int sl = 0; sl < NSLICES; ++sl) {
    char* cur = bsl + (sl & 1) * SLICEBYTES;
    char* nxt = bsl + ((sl & 1) ^ 1) * SLICEBYTES;
    const bool more = (sl + 1 < NSLICES);

    if (more) cg_stage_slice(Bt, nxt, tid, (sl + 1) * KSLICE);  // async fill

    V avn;
    if (active && more) {                       // prefetch next A chunk
      const _Float16* an = arow + (sl + 1) * KSLICE;
      avn.q[0] = *(const uint4*)(an);
      avn.q[1] = *(const uint4*)(an + 16);
    }

    if (active) {                               // wave-uniform guard
      // Pipeline B fragments one wmma ahead.
      V bv;
      bv.q[0] = *(const uint4*)(cur + bcol);
      bv.q[1] = *(const uint4*)(cur + bcol + 32);
#pragma unroll
      for (int r = 0; r < 8; ++r) {
        V bn;
        if (r < 7) {
          const char* l = cur + (r + 1) * 32 * COLSTRIDE + bcol;
          bn.q[0] = *(const uint4*)(l);
          bn.q[1] = *(const uint4*)(l + 32);
        }
        acc[r] = __builtin_amdgcn_wmma_f32_16x16x32_f16(
            false, av.h, false, bv.h, (short)0, acc[r], false, false);
        if (r < 7) bv = bn;
      }
    }

    cg_wait_async();                            // staging of nxt landed
    __syncthreads();                            // all reads of cur done
    if (more) av = avn;
  }

  if (active) {
    // Epilogue: max(0, acc[0..7]) elementwise == max_r relu.
    v8f mx = acc[0];
#pragma unroll
    for (int i = 0; i < 8; ++i) mx[i] = fmaxf(mx[i], 0.0f);
#pragma unroll
    for (int r = 1; r < 8; ++r)
#pragma unroll
      for (int i = 0; i < 8; ++i) mx[i] = fmaxf(mx[i], acc[r][i]);

    // C layout: lane l, elem i -> M = i + 8*lhalf, N = lmod.
    const int mbase = mtile * 16 + lhalf * 8;
#pragma unroll
    for (int i = 0; i < 8; ++i) {
      int m = mbase + i;
      if (m < mv) out[(size_t)m * N_OUT + osub + lmod] = mx[i];
    }
  }
}

// ---------------------------------------------------------------------------
extern "C" void kernel_launch(void* const* d_in, const int* in_sizes, int n_in,
                              void* d_out, int out_size, void* d_ws, size_t ws_size,
                              hipStream_t stream) {
  const float* sig  = (const float*)d_in[0];   // (M, 32) f32
  const float* kern = (const float*)d_in[1];   // (5, 8, 32, 32) f32
  const float* w    = (const float*)d_in[2];   // (M, 40, 3) f32
  const int*   idx  = (const int*)d_in[3];     // (M, 40, 3) i32
  const int*   rad  = (const int*)d_in[4];     // (M, 40) i32
  const int*   ang  = (const int*)d_in[5];     // (M, 40) i32
  float* out = (float*)d_out;                  // (M, 32) f32

  const int mv     = in_sizes[0] / N_IN;       // 50000
  const int mtiles = (mv + 15) / 16;

  _Float16* X = (_Float16*)d_ws;
  size_t xbytes = (size_t)mtiles * 16 * KDIM * sizeof(_Float16);
  xbytes = (xbytes + 255) & ~(size_t)255;
  _Float16* Bt = (_Float16*)((char*)d_ws + xbytes);

  cg_build_bt<<<(NCOL * KDIM + 255) / 256, 256, 0, stream>>>(kern, Bt);
  cg_pullback<<<(mv + 7) / 8, 256, 0, stream>>>(sig, w, idx, rad, ang, X, mv);
  cg_gemm_rot_max<<<(mtiles + 3) / 4, 256, 0, stream>>>(X, Bt, out, mtiles, mv);
}
